// NexusGNN_25331717111854
// MI455X (gfx1250) — matrix operand order repaired and checked by reading, not compiled, not run on hardware
//
#include <hip/hip_runtime.h>
#include <hip/hip_bf16.h>

typedef __attribute__((ext_vector_type(16))) _Float16 v16h;
typedef __attribute__((ext_vector_type(8)))  float    v8f;

// ---------------------------------------------------------------------------
// Utility: grid-stride zero fill
// ---------------------------------------------------------------------------
__global__ void zero_kernel(float* __restrict__ p, long long count) {
    long long i = (long long)blockIdx.x * blockDim.x + threadIdx.x;
    long long stride = (long long)gridDim.x * blockDim.x;
    for (; i < count; i += stride) p[i] = 0.0f;
}

// ---------------------------------------------------------------------------
// Degree accumulation over edges (self-loops added analytically later)
// ---------------------------------------------------------------------------
__global__ void degree_kernel(const int* __restrict__ dst, float* __restrict__ deg, int nE) {
    int e = blockIdx.x * blockDim.x + threadIdx.x;
    if (e < nE) atomicAdd(&deg[dst[e]], 1.0f);
}

// deg -> deg_inv_sqrt in place; +1 for the self-loop (so deg>0 always)
__global__ void rsqrt_kernel(float* __restrict__ deg, int n) {
    int i = blockIdx.x * blockDim.x + threadIdx.x;
    if (i < n) deg[i] = rsqrtf(deg[i] + 1.0f);
}

// ---------------------------------------------------------------------------
// WMMA GEMM:  C[nrows, NOUT] = A[nrows, K] @ W[K, NOUT]   (f32 in, f16 WMMA,
// f32 accumulate).  One wave per 16-row slab; NOUT/16 column tiles per wave.
// nrows must be a multiple of 16 (100000 = 6250*16).  Row guard is wave-
// uniform so EXEC is all-ones at every WMMA (ISA requirement).
// ---------------------------------------------------------------------------
template<int K, int NOUT>
__global__ void gemm_wmma_kernel(const float* __restrict__ A,
                                 const float* __restrict__ W,
                                 float* __restrict__ C,
                                 int nrows) {
    const int lane = threadIdx.x & 31;
    const int wave = threadIdx.x >> 5;
    const int row0 = (blockIdx.x * (blockDim.x >> 5) + wave) * 16;
    if (row0 >= nrows) return;               // uniform per wave

    const int hsel = lane >> 4;              // 0: lanes 0-15, 1: lanes 16-31
    const int lan  = lane & 15;

    constexpr int NT = NOUT / 16;
    v8f acc[NT];
#pragma unroll
    for (int t = 0; t < NT; ++t)
#pragma unroll
        for (int r = 0; r < 8; ++r) acc[t][r] = 0.0f;

    for (int kk = 0; kk < K; kk += 32) {
        // ---- A fragment: row = row0+lan; half idx g*8+o -> k = kk + g*16 + hsel*8 + o
        v16h a;
        const float* arow = A + (size_t)(row0 + lan) * K + kk + hsel * 8;
#pragma unroll
        for (int g = 0; g < 2; ++g)
#pragma unroll
            for (int o = 0; o < 8; ++o)
                a[g * 8 + o] = (_Float16)arow[g * 16 + o];

        // ---- B fragments: col = t*16+lan; half idx i -> k = kk + (i>>3)*16 + hsel*8 + (i&7)
        const float* wbase = W + (size_t)(kk + hsel * 8) * NOUT;
#pragma unroll
        for (int t = 0; t < NT; ++t) {
            v16h b;
            const float* wcol = wbase + t * 16 + lan;
#pragma unroll
            for (int i = 0; i < 16; ++i)
                b[i] = (_Float16)wcol[(size_t)((i & 7) + (i >> 3) * 16) * NOUT];

            acc[t] = __builtin_amdgcn_wmma_f32_16x16x32_f16(
                false, a, false, b, (short)0, acc[t], false, false);
        }
    }

    // ---- store: vgpr r -> row = row0 + r + 8*hsel, col = t*16 + lan
#pragma unroll
    for (int t = 0; t < NT; ++t) {
        float* crow = C + (size_t)(row0 + hsel * 8) * NOUT + t * 16 + lan;
#pragma unroll
        for (int r = 0; r < 8; ++r)
            crow[(size_t)r * NOUT] = acc[t][r];
    }
}

// ---------------------------------------------------------------------------
// Edge scatter: out[dst] += h[src] * dis[src]*dis[dst].  One thread per edge,
// float4 vector loads of the feature row; per-feature f32 atomics into an
// L2-resident output buffer.
// ---------------------------------------------------------------------------
template<int F>
__global__ void scatter_kernel(const float* __restrict__ h,
                               const int* __restrict__ src,
                               const int* __restrict__ dst,
                               const float* __restrict__ dis,
                               float* __restrict__ out, int nE) {
    int e = blockIdx.x * blockDim.x + threadIdx.x;
    if (e >= nE) return;
    const int s = src[e];
    const int d = dst[e];
    const float norm = dis[s] * dis[d];
    const float4* hs = (const float4*)(h + (size_t)s * F);
    float* od = out + (size_t)d * F;
#pragma unroll
    for (int i = 0; i < F / 4; ++i) {
        float4 v = hs[i];
        atomicAdd(od + 4 * i + 0, v.x * norm);
        atomicAdd(od + 4 * i + 1, v.y * norm);
        atomicAdd(od + 4 * i + 2, v.z * norm);
        atomicAdd(od + 4 * i + 3, v.w * norm);
    }
}

// ---------------------------------------------------------------------------
// Finalize: out[i,f] = (out[i,f] + h[i,f]*dis[i]^2 + bias[f]) [relu]
// (self-loop contribution folded in without atomics)
// ---------------------------------------------------------------------------
template<int F, bool RELU>
__global__ void finalize_kernel(float* __restrict__ out,
                                const float* __restrict__ h,
                                const float* __restrict__ dis,
                                const float* __restrict__ bias, int n) {
    long long idx = (long long)blockIdx.x * blockDim.x + threadIdx.x;
    int node = (int)(idx / F);
    int f    = (int)(idx % F);
    if (node >= n) return;
    float sn = dis[node];
    sn = sn * sn;
    size_t o = (size_t)node * F + f;
    float v = out[o] + h[o] * sn + bias[f];
    if (RELU) v = fmaxf(v, 0.0f);
    out[o] = v;
}

// ---------------------------------------------------------------------------
// kernel_launch
// Inputs (reference order): x[N,128], edge_index[2,E], W1[128,32], b1[32],
//                           W2[32,64], b2[64].  Output: [N,64] f32.
// ---------------------------------------------------------------------------
extern "C" void kernel_launch(void* const* d_in, const int* in_sizes, int n_in,
                              void* d_out, int out_size, void* d_ws, size_t ws_size,
                              hipStream_t stream) {
    const float* x    = (const float*)d_in[0];
    const int*   ei   = (const int*)d_in[1];
    const float* W1   = (const float*)d_in[2];
    const float* b1   = (const float*)d_in[3];
    const float* W2   = (const float*)d_in[4];
    const float* b2   = (const float*)d_in[5];
    float* out        = (float*)d_out;

    const int n  = in_sizes[0] / 128;   // 100000 nodes
    const int nE = in_sizes[1] / 2;     // 3.2M edges
    const int* src = ei;
    const int* dst = ei + nE;

    // workspace layout (floats): dis[n] | h1[n*32] | agg1[n*32] | h2[n*64]
    float* dis  = (float*)d_ws;
    float* h1   = dis  + n;
    float* agg1 = h1   + (size_t)n * 32;
    float* h2   = agg1 + (size_t)n * 32;

    const int B = 256;
    const int zgrid = 2048;  // grid-stride zero fill

    // 1) degree -> deg_inv_sqrt
    zero_kernel<<<zgrid, B, 0, stream>>>(dis, (long long)n);
    degree_kernel<<<(nE + B - 1) / B, B, 0, stream>>>(dst, dis, nE);
    rsqrt_kernel<<<(n + B - 1) / B, B, 0, stream>>>(dis, n);

    // 2) layer 1: h1 = x @ W1   (WMMA)
    {
        const int rows_per_block = (B / 32) * 16;  // 128
        const int grid = (n + rows_per_block - 1) / rows_per_block;
        gemm_wmma_kernel<128, 32><<<grid, B, 0, stream>>>(x, W1, h1, n);
    }

    // 3) scatter layer 1 into agg1, finalize (+self-loop, +b1, relu)
    zero_kernel<<<zgrid, B, 0, stream>>>(agg1, (long long)n * 32);
    scatter_kernel<32><<<(nE + B - 1) / B, B, 0, stream>>>(h1, src, dst, dis, agg1, nE);
    {
        long long tot = (long long)n * 32;
        finalize_kernel<32, true><<<(unsigned)((tot + B - 1) / B), B, 0, stream>>>(agg1, h1, dis, b1, n);
    }

    // 4) layer 2: h2 = agg1 @ W2   (WMMA)
    {
        const int rows_per_block = (B / 32) * 16;  // 128
        const int grid = (n + rows_per_block - 1) / rows_per_block;
        gemm_wmma_kernel<32, 64><<<grid, B, 0, stream>>>(agg1, W2, h2, n);
    }

    // 5) scatter layer 2 into d_out, finalize (+self-loop, +b2)
    zero_kernel<<<zgrid, B, 0, stream>>>(out, (long long)n * 64);
    scatter_kernel<64><<<(nE + B - 1) / B, B, 0, stream>>>(h2, src, dst, dis, out, nE);
    {
        long long tot = (long long)n * 64;
        finalize_kernel<64, false><<<(unsigned)((tot + B - 1) / B), B, 0, stream>>>(out, h2, dis, b2, n);
    }
}
